// TransformerEncoder_66374424592875
// MI455X (gfx1250) — compile-verified, matrix operand order checked by reading, not run on hardware
//
#include <hip/hip_runtime.h>
#include <math.h>

// ---------------- problem constants ----------------
#define BB      8
#define SEQ     1024
#define DIN     64
#define DMODEL  512
#define NHEAD   8
#define HDIM    64
#define NLAYER  4
#define TOKENS  (BB * SEQ)          // 8192
#define QKVW    (3 * DMODEL)        // 1536
#define FFW     (4 * DMODEL)        // 2048

// ---------------- WMMA vector types ----------------
typedef __attribute__((ext_vector_type(16))) __bf16 v16bf;
typedef __attribute__((ext_vector_type(8)))  __bf16 v8bf;
typedef __attribute__((ext_vector_type(8)))  float  v8f;

__device__ __forceinline__ unsigned short f2bf(float f) {
    unsigned u = __float_as_uint(f);
    u += 0x7FFFu + ((u >> 16) & 1u);     // round-to-nearest-even
    return (unsigned short)(u >> 16);
}
__device__ __forceinline__ float bf2f(unsigned short h) {
    return __uint_as_float(((unsigned)h) << 16);
}

// Load one 16x32 bf16 WMMA operand fragment (16 bf16 per lane) from a
// row-major (rows x ldk) matrix.  Caller passes the per-lane base pointer
//   &M[row * ldk + kBase + koff]   with koff = (lane>=16 ? 8 : 0).
// Per the CDNA5 ISA layout, each lane holds K = koff+{0..7} and koff+{16..23}:
// two contiguous 16-byte chunks.
__device__ __forceinline__ v16bf load_frag(const unsigned short* p) {
    v8bf lo = *(const v8bf*)(p);
    v8bf hi = *(const v8bf*)(p + 16);
    return __builtin_shufflevector(lo, hi, 0,1,2,3,4,5,6,7,8,9,10,11,12,13,14,15);
}

#define WMMA_BF16(A, Bx, C) \
    __builtin_amdgcn_wmma_f32_16x16x32_bf16(false, (A), false, (Bx), (short)0, (C), false, false)

// ---------------- small helper kernels ----------------
__global__ void zero_kernel(float* __restrict__ p, int n) {
    int i = blockIdx.x * 256 + threadIdx.x;
    if (i < n) p[i] = 0.0f;
}

__global__ void convert_bf16_kernel(const float* __restrict__ in,
                                    unsigned short* __restrict__ out, int n) {
    int i = blockIdx.x * 256 + threadIdx.x;
    if (i < n) out[i] = f2bf(in[i]);
}

// x_out[token, d] = x_in[token,:] . W[d,:] + b[d] + posemb(token%S, d)
__global__ void __launch_bounds__(256) input_proj_kernel(
    const float* __restrict__ xin, const float* __restrict__ w,
    const float* __restrict__ bias, float* __restrict__ xout) {
    size_t idx = (size_t)blockIdx.x * 256 + threadIdx.x;   // TOKENS*DMODEL total
    int d       = (int)(idx & (DMODEL - 1));
    size_t row  = idx >> 9;                                // /DMODEL
    const float* xr = xin + row * DIN;
    const float* wr = w + (size_t)d * DIN;
    float acc = bias[d];
#pragma unroll 8
    for (int k = 0; k < DIN; ++k) acc += xr[k] * wr[k];
    int s  = (int)(row & (SEQ - 1));
    int i2 = d & ~1;
    float div = expf(-(float)i2 * (9.210340371976184f / (float)DMODEL)); // ln(1e4)/D
    float ang = (float)s * div;
    acc += (d & 1) ? cosf(ang) : sinf(ang);
    xout[idx] = acc;
}

// one wave32 per 512-wide row; bf16 output
__global__ void __launch_bounds__(256) ln_kernel(
    const float* __restrict__ x, const float* __restrict__ scale,
    const float* __restrict__ bias, unsigned short* __restrict__ out) {
    const int lane = threadIdx.x & 31;
    const int wave = threadIdx.x >> 5;
    const size_t row = (size_t)blockIdx.x * 8 + wave;
    const float* xr = x + row * DMODEL;
    float v[16];
    float s = 0.0f;
#pragma unroll
    for (int u = 0; u < 16; ++u) { v[u] = xr[lane + u * 32]; s += v[u]; }
#pragma unroll
    for (int off = 1; off < 32; off <<= 1) s += __shfl_xor(s, off, 32);
    float mean = s * (1.0f / (float)DMODEL);
    float q = 0.0f;
#pragma unroll
    for (int u = 0; u < 16; ++u) { float d = v[u] - mean; q += d * d; }
#pragma unroll
    for (int off = 1; off < 32; off <<= 1) q += __shfl_xor(q, off, 32);
    float inv = rsqrtf(q * (1.0f / (float)DMODEL) + 1e-5f);
#pragma unroll
    for (int u = 0; u < 16; ++u) {
        int d = lane + u * 32;
        out[row * DMODEL + d] = f2bf((v[u] - mean) * inv * scale[d] + bias[d]);
    }
}

// ---------------- WMMA GEMM:  C = A(MxK,bf16) . W(NxK,bf16)^T + bias ----------------
// 256 threads = 8 waves arranged 2(M) x 4(N); block tile 128 x 256;
// wave tile 64 x 64 = 4x4 WMMA accumulators -> 16 WMMA per 8 fragment loads.
// Epilogue staged through wave-private LDS for coalesced b128 global traffic.
enum { EP_STORE = 0, EP_GELU = 1, EP_RES = 2 };

template <int EP>
__global__ void __launch_bounds__(256) gemm_bf16_kernel(
    const unsigned short* __restrict__ A,   // M x K bf16
    const unsigned short* __restrict__ W,   // N x K bf16 (acts as B^T)
    const float* __restrict__ bias,         // N
    unsigned short* __restrict__ outBf,     // M x N bf16 (EP_STORE / EP_GELU)
    float* __restrict__ resid,              // M x N fp32 (EP_RES: +=)
    int M, int N, int K) {
    (void)M;
    __shared__ alignas(16) float sEpi[8][16 * 64];   // 4KB per wave, wave-private

    const int lane  = threadIdx.x & 31;
    const int wave  = threadIdx.x >> 5;
    const int fr    = lane & 15;
    const int koff  = (lane >> 4) * 8;
    const int mHalf = lane >> 4;

    const int gm0 = blockIdx.y * 128 + (wave & 1) * 64;
    const int gn0 = blockIdx.x * 256 + (wave >> 1) * 64;

    v8f acc[4][4];
#pragma unroll
    for (int mi = 0; mi < 4; ++mi)
#pragma unroll
        for (int ni = 0; ni < 4; ++ni) acc[mi][ni] = (v8f){};

    const unsigned short* aBase = A + (size_t)(gm0 + fr) * K + koff;
    const unsigned short* wBase = W + (size_t)(gn0 + fr) * K + koff;

    for (int k0 = 0; k0 < K; k0 += 32) {
        __builtin_prefetch(aBase + k0 + 64, 0, 0);   // global_prefetch_b8
        __builtin_prefetch(wBase + k0 + 64, 0, 0);
        v16bf a[4];
#pragma unroll
        for (int mi = 0; mi < 4; ++mi)
            a[mi] = load_frag(aBase + (size_t)(mi * 16) * K + k0);
#pragma unroll
        for (int ni = 0; ni < 4; ++ni) {
            v16bf b = load_frag(wBase + (size_t)(ni * 16) * K + k0);
#pragma unroll
            for (int mi = 0; mi < 4; ++mi)
                acc[mi][ni] = WMMA_BF16(a[mi], b, acc[mi][ni]);
        }
    }

    float* myF = sEpi[wave];
    unsigned short* myH = (unsigned short*)myF;

#pragma unroll
    for (int mi = 0; mi < 4; ++mi) {
        if (EP == EP_RES) {
            // stage fp32 sub-tile, then coalesced float4 read-modify-write
#pragma unroll
            for (int ni = 0; ni < 4; ++ni) {
                float bv = bias[gn0 + ni * 16 + fr];
#pragma unroll
                for (int e = 0; e < 8; ++e)
                    myF[(mHalf * 8 + e) * 64 + ni * 16 + fr] = acc[mi][ni][e] + bv;
            }
#pragma unroll
            for (int p = 0; p < 8; ++p) {
                int el = p * 128 + lane * 4;
                int mr = el >> 6, nc = el & 63;
                float4 c = *(const float4*)&myF[el];
                float* g = &resid[(size_t)(gm0 + mi * 16 + mr) * N + gn0 + nc];
                float4 r = *(const float4*)g;
                r.x += c.x; r.y += c.y; r.z += c.z; r.w += c.w;
                *(float4*)g = r;
            }
        } else {
            // stage bf16 sub-tile, then coalesced b128 stores
#pragma unroll
            for (int ni = 0; ni < 4; ++ni) {
                float bv = bias[gn0 + ni * 16 + fr];
#pragma unroll
                for (int e = 0; e < 8; ++e) {
                    float c = acc[mi][ni][e] + bv;
                    if (EP == EP_GELU) c = 0.5f * c * (1.0f + erff(c * 0.70710678118654752f));
                    myH[(mHalf * 8 + e) * 64 + ni * 16 + fr] = f2bf(c);
                }
            }
#pragma unroll
            for (int p = 0; p < 4; ++p) {
                int el = p * 256 + lane * 8;
                int mr = el >> 6, nc = el & 63;
                v8bf v = *(const v8bf*)&myH[el];
                *(v8bf*)&outBf[(size_t)(gm0 + mi * 16 + mr) * N + gn0 + nc] = v;
            }
        }
    }
}

// ---------------- flash-style causal attention + entropy ----------------
// grid (S/64, NHEAD, BB), 256 threads = 8 waves.
__global__ void __launch_bounds__(256) attn_kernel(
    const unsigned short* __restrict__ qkv,   // TOKENS x 1536 bf16 [q|k|v]
    unsigned short* __restrict__ attnOut,     // TOKENS x 512 bf16
    float* __restrict__ entOut) {             // NHEAD floats (this layer)
    const int qb = blockIdx.x;   // query block (64 rows)
    const int h  = blockIdx.y;
    const int b  = blockIdx.z;

    __shared__ alignas(16) float          sS[64 * 64];    // fp32 scores
    __shared__ alignas(16) unsigned short sP[64 * 64];    // bf16 probs
    __shared__ alignas(16) unsigned short sVT[64 * 64];   // bf16 V^T  [hd][key]
    __shared__ float rowM[64], rowZ[64], rowT[64], rowScale[64];

    const int lane  = threadIdx.x & 31;
    const int wave  = threadIdx.x >> 5;
    const int fr    = lane & 15;
    const int koff  = (lane >> 4) * 8;
    const int mHalf = lane >> 4;
    const int lq    = (wave & 3) * 16;   // local query sub-tile
    const int sub2  = wave >> 2;         // 0/1: key half (scores) / hd half (output)

    const size_t baseTok = (size_t)b * SEQ;
    const size_t LD = QKVW;              // 1536

    if (threadIdx.x < 64) {
        rowM[threadIdx.x] = -3.0e38f; rowZ[threadIdx.x] = 0.0f;
        rowT[threadIdx.x] = 0.0f;     rowScale[threadIdx.x] = 0.0f;
    }

    // preload Q A-fragments (reused for every key block)
    v16bf aQ[2];
#pragma unroll
    for (int ks = 0; ks < 2; ++ks)
        aQ[ks] = load_frag(qkv + (baseTok + qb * 64 + lq + fr) * LD + h * HDIM + ks * 32 + koff);

    v8f accO[2]; accO[0] = (v8f){}; accO[1] = (v8f){};
    __syncthreads();

    for (int kb = 0; kb <= qb; ++kb) {
        // ---- stage V^T into LDS ----
        {
            int j  = threadIdx.x >> 2;          // key row 0..63
            int hc = (threadIdx.x & 3) * 16;    // hd chunk
            const unsigned short* vp =
                qkv + (baseTok + kb * 64 + j) * LD + 2 * DMODEL + h * HDIM + hc;
#pragma unroll
            for (int u = 0; u < 16; ++u) sVT[(hc + u) * 64 + j] = vp[u];
        }
        // ---- scores: S = Q K^T / sqrt(HD) ----
#pragma unroll
        for (int sub = 0; sub < 2; ++sub) {
            v8f sAcc = (v8f){};
            int keyRow = kb * 64 + sub2 * 32 + sub * 16 + fr;
            v16bf bk0 = load_frag(qkv + (baseTok + keyRow) * LD + DMODEL + h * HDIM +  0 + koff);
            v16bf bk1 = load_frag(qkv + (baseTok + keyRow) * LD + DMODEL + h * HDIM + 32 + koff);
            sAcc = WMMA_BF16(aQ[0], bk0, sAcc);
            sAcc = WMMA_BF16(aQ[1], bk1, sAcc);
            int lk = sub2 * 32 + sub * 16;
#pragma unroll
            for (int e = 0; e < 8; ++e)
                sS[(lq + mHalf * 8 + e) * 64 + lk + fr] = sAcc[e] * 0.125f;
        }
        __syncthreads();
        // ---- online softmax pass (4 threads per row, 16 keys each) ----
        {
            int r  = threadIdx.x >> 2;
            int c0 = (threadIdx.x & 3) * 16;
            int gq = qb * 64 + r;
            float oldM = rowM[r];
            float vals[16];
            bool  msk[16];
            float pm = -3.0e38f;
#pragma unroll
            for (int u = 0; u < 16; ++u) {
                int key = kb * 64 + c0 + u;
                float v = sS[r * 64 + c0 + u];
                bool m = key > gq;
                msk[u] = m; vals[u] = v;
                if (!m) pm = fmaxf(pm, v);
            }
#pragma unroll
            for (int off = 1; off < 4; off <<= 1) pm = fmaxf(pm, __shfl_xor(pm, off, 32));
            float newM = fmaxf(oldM, pm);
            float pz = 0.0f, pt = 0.0f;
#pragma unroll
            for (int u = 0; u < 16; ++u) {
                float e = msk[u] ? 0.0f : expf(vals[u] - newM);
                pz += e; pt += e * vals[u];
                sP[r * 64 + c0 + u] = f2bf(e);
            }
#pragma unroll
            for (int off = 1; off < 4; off <<= 1) {
                pz += __shfl_xor(pz, off, 32);
                pt += __shfl_xor(pt, off, 32);
            }
            if ((threadIdx.x & 3) == 0) {
                float sc = expf(oldM - newM);
                rowScale[r] = sc;
                rowZ[r] = rowZ[r] * sc + pz;
                rowT[r] = rowT[r] * sc + pt;
                rowM[r] = newM;
            }
        }
        __syncthreads();
        // ---- rescale running O, then O += P V via WMMA from LDS ----
#pragma unroll
        for (int sub = 0; sub < 2; ++sub)
#pragma unroll
            for (int e = 0; e < 8; ++e) accO[sub][e] *= rowScale[lq + mHalf * 8 + e];
#pragma unroll
        for (int ks = 0; ks < 2; ++ks) {
            v16bf aP = load_frag(&sP[(lq + fr) * 64 + ks * 32 + koff]);
#pragma unroll
            for (int sub = 0; sub < 2; ++sub) {
                v16bf bV = load_frag(&sVT[(sub2 * 32 + sub * 16 + fr) * 64 + ks * 32 + koff]);
                accO[sub] = WMMA_BF16(aP, bV, accO[sub]);
            }
        }
        __syncthreads();
    }

    // ---- epilogue: normalize, write bf16 attn output ----
#pragma unroll
    for (int sub = 0; sub < 2; ++sub)
#pragma unroll
        for (int e = 0; e < 8; ++e) {
            int lr = lq + mHalf * 8 + e;
            int gq = qb * 64 + lr;
            float val = accO[sub][e] / rowZ[lr];
            attnOut[(baseTok + gq) * DMODEL + h * HDIM + sub2 * 32 + sub * 16 + fr] = f2bf(val);
        }
    // ---- entropy: H = m + log Z - T/Z, mean over (B, S) ----
    if (threadIdx.x < 64) {
        int r = threadIdx.x;
        float Z = rowZ[r];
        float Hent = rowM[r] + logf(Z) - rowT[r] / Z;
        atomicAdd(&entOut[h], Hent * (1.0f / (float)TOKENS));
    }
}

// ---------------- launcher ----------------
extern "C" void kernel_launch(void* const* d_in, const int* in_sizes, int n_in,
                              void* d_out, int out_size, void* d_ws, size_t ws_size,
                              hipStream_t stream) {
    (void)in_sizes; (void)n_in; (void)out_size; (void)ws_size;
    const float* x_in = (const float*)d_in[0];
    const float* ipw  = (const float*)d_in[1];
    const float* ipb  = (const float*)d_in[2];
    const float* ln1s = (const float*)d_in[3];
    const float* ln1b = (const float*)d_in[4];
    const float* wqkv = (const float*)d_in[5];
    const float* bqkv = (const float*)d_in[6];
    const float* wout = (const float*)d_in[7];
    const float* bout = (const float*)d_in[8];
    const float* ln2s = (const float*)d_in[9];
    const float* ln2b = (const float*)d_in[10];
    const float* w1   = (const float*)d_in[11];
    const float* b1   = (const float*)d_in[12];
    const float* w2   = (const float*)d_in[13];
    const float* b2   = (const float*)d_in[14];

    float* x   = (float*)d_out;                      // (B,S,D) residual stream
    float* ent = x + (size_t)TOKENS * DMODEL;        // (NL,H) entropies

    // workspace carve-up (bf16 elements)
    unsigned short* ws = (unsigned short*)d_ws;
    size_t off = 0;
    const size_t NWQ = (size_t)NLAYER * QKVW * DMODEL;
    const size_t NWO = (size_t)NLAYER * DMODEL * DMODEL;
    const size_t NW1 = (size_t)NLAYER * FFW * DMODEL;
    const size_t NW2 = (size_t)NLAYER * DMODEL * FFW;
    unsigned short* wq_b = ws + off; off += NWQ;
    unsigned short* wo_b = ws + off; off += NWO;
    unsigned short* w1_b = ws + off; off += NW1;
    unsigned short* w2_b = ws + off; off += NW2;
    unsigned short* hbuf = ws + off; off += (size_t)TOKENS * DMODEL;
    unsigned short* qkvb = ws + off; off += (size_t)TOKENS * QKVW;
    unsigned short* aob  = ws + off; off += (size_t)TOKENS * DMODEL;
    unsigned short* mhb  = ws + off; off += (size_t)TOKENS * FFW;

    zero_kernel<<<1, 256, 0, stream>>>(ent, NLAYER * NHEAD);
    convert_bf16_kernel<<<(int)((NWQ + 255) / 256), 256, 0, stream>>>(wqkv, wq_b, (int)NWQ);
    convert_bf16_kernel<<<(int)((NWO + 255) / 256), 256, 0, stream>>>(wout, wo_b, (int)NWO);
    convert_bf16_kernel<<<(int)((NW1 + 255) / 256), 256, 0, stream>>>(w1, w1_b, (int)NW1);
    convert_bf16_kernel<<<(int)((NW2 + 255) / 256), 256, 0, stream>>>(w2, w2_b, (int)NW2);

    input_proj_kernel<<<(TOKENS * DMODEL) / 256, 256, 0, stream>>>(x_in, ipw, ipb, x);

    for (int l = 0; l < NLAYER; ++l) {
        ln_kernel<<<TOKENS / 8, 256, 0, stream>>>(x, ln1s + l * DMODEL, ln1b + l * DMODEL, hbuf);
        gemm_bf16_kernel<EP_STORE><<<dim3(QKVW / 256, TOKENS / 128), 256, 0, stream>>>(
            hbuf, wq_b + (size_t)l * QKVW * DMODEL, bqkv + (size_t)l * QKVW,
            qkvb, nullptr, TOKENS, QKVW, DMODEL);
        attn_kernel<<<dim3(SEQ / 64, NHEAD, BB), 256, 0, stream>>>(qkvb, aob, ent + l * NHEAD);
        gemm_bf16_kernel<EP_RES><<<dim3(DMODEL / 256, TOKENS / 128), 256, 0, stream>>>(
            aob, wo_b + (size_t)l * DMODEL * DMODEL, bout + (size_t)l * DMODEL,
            nullptr, x, TOKENS, DMODEL, DMODEL);
        ln_kernel<<<TOKENS / 8, 256, 0, stream>>>(x, ln2s + l * DMODEL, ln2b + l * DMODEL, hbuf);
        gemm_bf16_kernel<EP_GELU><<<dim3(FFW / 256, TOKENS / 128), 256, 0, stream>>>(
            hbuf, w1_b + (size_t)l * FFW * DMODEL, b1 + (size_t)l * FFW,
            mhb, nullptr, TOKENS, FFW, DMODEL);
        gemm_bf16_kernel<EP_RES><<<dim3(DMODEL / 256, TOKENS / 128), 256, 0, stream>>>(
            mhb, w2_b + (size_t)l * DMODEL * FFW, b2 + (size_t)l * DMODEL,
            nullptr, x, TOKENS, DMODEL, FFW);
    }
}
